// FastBaseAttention_49916109914157
// MI455X (gfx1250) — matrix operand order repaired and checked by reading, not compiled, not run on hardware
//
#include <hip/hip_runtime.h>

// RoPE + sliding-window causal flash attention for MI455X (gfx1250, wave32).
// B=2, S=2048, H=16, D=64, WINDOW=256. fp32 in/out, f16 WMMA compute, f32 acc.
// K/V stream: double-buffered GLOBAL_LOAD_ASYNC_TO_LDS (ASYNCcnt pipelined).

typedef __attribute__((ext_vector_type(16))) _Float16 v16h;
typedef __attribute__((ext_vector_type(8)))  _Float16 v8h;
typedef __attribute__((ext_vector_type(8)))  float    v8f;
typedef __attribute__((vector_size(16)))     int      b128i;  // builtin's pointee

#define B_    2
#define S_    2048
#define H_    16
#define D_    64
#define WIN   256
#define QBLK  64                    // queries per workgroup (4 waves x 16)
#define KT    32                    // key tile size
#define NTIL  ((QBLK + WIN) / KT)   // 10 key tiles per query block
#define NEGB  (-3.0e38f)

// ---- gfx1250 async global->LDS path (guarded; sync fallback keeps compile OK)
#if __has_builtin(__builtin_amdgcn_global_load_async_to_lds_b128)
#define HAS_ASYNC_LDS 1
#define ASYNC16(gp, lp, off)                                                   \
    __builtin_amdgcn_global_load_async_to_lds_b128(                            \
        (__attribute__((address_space(1))) b128i*)(gp),                        \
        (__attribute__((address_space(3))) b128i*)(lp), (off), 0)
#if __has_builtin(__builtin_amdgcn_s_wait_asynccnt)
#define WAIT_ASYNC(n) __builtin_amdgcn_s_wait_asynccnt(n)
#else
#define WAIT_ASYNC(n) asm volatile("s_wait_asynccnt %0" ::"i"(n) : "memory")
#endif
#else
#define HAS_ASYNC_LDS 0
#define ASYNC16(gp, lp, off)                                                   \
    do {                                                                       \
        *(float4*)((char*)(lp) + (off)) =                                      \
            *(const float4*)((const char*)(gp) + (off));                       \
    } while (0)
#define WAIT_ASYNC(n) do { } while (0)
#endif

__device__ __forceinline__ float inv_freq(int p) {
    // 10000^(-p/32) = exp2(-p * log2(10000)/32)
    return exp2f((float)p * -0.41524101186092864f);
}

__device__ __forceinline__ v16h cat8(v8h lo, v8h hi) {
    return __builtin_shufflevector(lo, hi, 0,1,2,3,4,5,6,7,8,9,10,11,12,13,14,15);
}

__global__ __launch_bounds__(128)
void fa_rope_swa_kernel(const float* __restrict__ q, const float* __restrict__ k,
                        const float* __restrict__ v, float* __restrict__ out)
{
    __shared__ float    SgK[2][KT * D_];  // raw f32 K staging (double buffered)
    __shared__ float    SgV[2][KT * D_];  // raw f32 V staging (double buffered)
    __shared__ _Float16 Kl[KT * D_];      // K tile, row-major [kk][d], RoPE'd f16
    __shared__ _Float16 Vt[D_ * KT];      // V tile, transposed [d][kk], f16
    __shared__ _Float16 Ps[4][16 * KT];   // per-wave P scratch (C/D -> A layout)

    const int tid  = threadIdx.x;
    const int lane = tid & 31;
    const int wid  = tid >> 5;

    const int qblk = blockIdx.x % (S_ / QBLK);
    const int hh   = (blockIdx.x / (S_ / QBLK)) % H_;
    const int bb   = blockIdx.x / ((S_ / QBLK) * H_);

    const int q0     = qblk * QBLK;
    const int qw     = q0 + wid * 16;    // this wave's 16-query tile base
    const int kstart = q0 - WIN;         // first key needed by the block

    const int half  = (lane < 16) ? 0 : 1;
    const int n16   = lane & 15;         // column / A-row index for this lane
    const int base8 = half * 8;          // A-layout K-group select
    const int kb16  = half * 16;         // unused; B operands also take base8

    // staging geometry: each thread owns 16 consecutive floats of one key row
    const int srow = tid >> 2;           // key row within tile, 0..31
    const int sd16 = (tid & 3) * 16;     // d offset 0,16,32,48
    const size_t bh_off = ((size_t)bb * S_) * H_ + hh;  // row j -> (bh_off + j*H_)*D_

    // ---------------- Q load + RoPE into A-operand registers ----------------
    // A (16x32 f16) lane layout: M = lane&15; halves 0..7 -> K=base8+h,
    // halves 8..15 -> K=16+base8+(h-8). Two chunks cover D=64.
    const int iq = qw + n16;
    const float* qp = q + (((size_t)bb * S_ + iq) * H_ + hh) * D_;
    float g[4][8];  // d groups: base8 + {0,16,32,48} + [0..8)
#pragma unroll
    for (int gi = 0; gi < 4; ++gi) {
        const float4 lo = *(const float4*)(qp + base8 + 16 * gi);
        const float4 hi = *(const float4*)(qp + base8 + 16 * gi + 4);
        g[gi][0]=lo.x; g[gi][1]=lo.y; g[gi][2]=lo.z; g[gi][3]=lo.w;
        g[gi][4]=hi.x; g[gi][5]=hi.y; g[gi][6]=hi.z; g[gi][7]=hi.w;
    }
    // RoPE pairs: (g0 <-> g2) at p=base8+i, (g1 <-> g3) at p=base8+16+i
    float r0[8], r1[8], r2[8], r3[8];
#pragma unroll
    for (int i = 0; i < 8; ++i) {
        {
            const float ang = (float)iq * inv_freq(base8 + i);
            const float c = __cosf(ang), s = __sinf(ang);
            r0[i] = g[0][i] * c - g[2][i] * s;
            r2[i] = g[2][i] * c + g[0][i] * s;
        }
        {
            const float ang = (float)iq * inv_freq(base8 + 16 + i);
            const float c = __cosf(ang), s = __sinf(ang);
            r1[i] = g[1][i] * c - g[3][i] * s;
            r3[i] = g[3][i] * c + g[1][i] * s;
        }
    }
    v16h qa[2];
#pragma unroll
    for (int i = 0; i < 8; ++i) {
        qa[0][i]     = (_Float16)r0[i];
        qa[0][i + 8] = (_Float16)r1[i];
        qa[1][i]     = (_Float16)r2[i];
        qa[1][i + 8] = (_Float16)r3[i];
    }

    // ---------------- flash-attention state ----------------
    // C/D layout: element (m, n) with n = lane&15, m = r + (lane<16 ? 0 : 8).
    // After xor-reductions over masks 1..8, lane stats align with rows base8+r.
    float mrow[8], lrow[8];
    v8f acc[4];
#pragma unroll
    for (int r = 0; r < 8; ++r) { mrow[r] = NEGB; lrow[r] = 0.0f; }
#pragma unroll
    for (int c = 0; c < 4; ++c)
#pragma unroll
        for (int e = 0; e < 8; ++e) acc[c][e] = 0.0f;

    const float scale = 0.125f;  // D^-0.5

    // ---------------- prologue: stage tile 0 asynchronously ----------------
    {
        int j = kstart + srow;
        j = j < 0 ? 0 : j;  // clamped rows are fully masked later
        const float* kp = k + (bh_off + (size_t)j * H_) * D_ + sd16;
        const float* vp = v + (bh_off + (size_t)j * H_) * D_ + sd16;
        float* sk = &SgK[0][srow * D_ + sd16];
        float* sv = &SgV[0][srow * D_ + sd16];
        ASYNC16(kp, sk, 0);  ASYNC16(kp, sk, 16);
        ASYNC16(kp, sk, 32); ASYNC16(kp, sk, 48);
        ASYNC16(vp, sv, 0);  ASYNC16(vp, sv, 16);
        ASYNC16(vp, sv, 32); ASYNC16(vp, sv, 48);
    }

    // ---------------- key-tile loop (double-buffered async stream) ----------
    for (int t = 0; t < NTIL; ++t) {
        const int ktb = kstart + t * KT;
        const int buf = t & 1;

        // stage tile t+1 into the other buffer while tile t is consumed
        if (t + 1 < NTIL) {
            int j = ktb + KT + srow;
            j = j < 0 ? 0 : j;
            const float* kp = k + (bh_off + (size_t)j * H_) * D_ + sd16;
            const float* vp = v + (bh_off + (size_t)j * H_) * D_ + sd16;
            float* sk = &SgK[buf ^ 1][srow * D_ + sd16];
            float* sv = &SgV[buf ^ 1][srow * D_ + sd16];
            ASYNC16(kp, sk, 0);  ASYNC16(kp, sk, 16);
            ASYNC16(kp, sk, 32); ASYNC16(kp, sk, 48);
            ASYNC16(vp, sv, 0);  ASYNC16(vp, sv, 16);
            ASYNC16(vp, sv, 32); ASYNC16(vp, sv, 48);
            WAIT_ASYNC(8);  // drain tile t, keep tile t+1's 8 ops in flight
        } else {
            WAIT_ASYNC(0);
        }
        __syncthreads();  // tile t staging visible to all waves

        // -- convert staging: K += RoPE -> f16 row-major; V -> f16 transposed
        {
            const int d0 = (tid & 3) * 8;           // 0,8,16,24 (pairs with +32)
            const int jr0 = ktb + srow;
            const int jr = jr0 < 0 ? 0 : jr0;       // rope angle for clamped row
            const float* skp = &SgK[buf][srow * D_];
            const float* svp = &SgV[buf][srow * D_];
            const float4 kl0 = *(const float4*)(skp + d0);
            const float4 kl1 = *(const float4*)(skp + d0 + 4);
            const float4 kh0 = *(const float4*)(skp + d0 + 32);
            const float4 kh1 = *(const float4*)(skp + d0 + 36);
            const float4 vl0 = *(const float4*)(svp + d0);
            const float4 vl1 = *(const float4*)(svp + d0 + 4);
            const float4 vh0 = *(const float4*)(svp + d0 + 32);
            const float4 vh1 = *(const float4*)(svp + d0 + 36);
            const float klo[8] = {kl0.x,kl0.y,kl0.z,kl0.w,kl1.x,kl1.y,kl1.z,kl1.w};
            const float khi[8] = {kh0.x,kh0.y,kh0.z,kh0.w,kh1.x,kh1.y,kh1.z,kh1.w};
            const float vlo[8] = {vl0.x,vl0.y,vl0.z,vl0.w,vl1.x,vl1.y,vl1.z,vl1.w};
            const float vhi[8] = {vh0.x,vh0.y,vh0.z,vh0.w,vh1.x,vh1.y,vh1.z,vh1.w};
            v8h klds, khds;
#pragma unroll
            for (int i = 0; i < 8; ++i) {
                const float ang = (float)jr * inv_freq(d0 + i);
                const float c = __cosf(ang), s = __sinf(ang);
                klds[i] = (_Float16)(klo[i] * c - khi[i] * s);
                khds[i] = (_Float16)(khi[i] * c + klo[i] * s);
            }
            *(v8h*)(&Kl[srow * D_ + d0])      = klds;
            *(v8h*)(&Kl[srow * D_ + d0 + 32]) = khds;
#pragma unroll
            for (int i = 0; i < 8; ++i) {
                Vt[(d0 + i)      * KT + srow] = (_Float16)vlo[i];
                Vt[(d0 + 32 + i) * KT + srow] = (_Float16)vhi[i];
            }
        }
        __syncthreads();  // Kl/Vt ready

        // wave-uniform skip of tiles fully outside this wave's window
        const bool active = (ktb + KT - 1 >= qw - WIN) && (ktb <= qw + 15);
        if (active) {
            // ---- scores S = Q K^T : two 16-col subtiles, K-dim split in 2 ----
            v8f sc[2];
#pragma unroll
            for (int nt = 0; nt < 2; ++nt)
#pragma unroll
                for (int e = 0; e < 8; ++e) sc[nt][e] = 0.0f;
#pragma unroll
            for (int nt = 0; nt < 2; ++nt) {
#pragma unroll
                for (int c = 0; c < 2; ++c) {
                    // B (32x16): lane col = key row nt*16+n16; d in A (table) order
                    const _Float16* bp = &Kl[(nt * 16 + n16) * D_ + 32 * c + base8];
                    const v8h blo = *(const v8h*)bp;
                    const v8h bhi = *(const v8h*)(bp + 16);
                    sc[nt] = __builtin_amdgcn_wmma_f32_16x16x32_f16(
                        false, qa[c], false, cat8(blo, bhi), (short)0, sc[nt],
                        false, false);
                }
            }

            // ---- mask + scale ----
#pragma unroll
            for (int nt = 0; nt < 2; ++nt) {
                const int j0 = ktb + nt * 16 + n16;
#pragma unroll
                for (int r = 0; r < 8; ++r) {
                    const int i_row = qw + base8 + r;
                    const bool ok = (j0 >= 0) && (j0 <= i_row) && (j0 >= i_row - WIN);
                    sc[nt][r] = ok ? sc[nt][r] * scale : NEGB;
                }
            }

            // ---- online softmax: row max/sum via 16-lane xor reductions ----
            float mnew[8];
#pragma unroll
            for (int r = 0; r < 8; ++r) {
                float vmx = fmaxf(sc[0][r], sc[1][r]);
#pragma unroll
                for (int off = 1; off < 16; off <<= 1)
                    vmx = fmaxf(vmx, __shfl_xor(vmx, off, 32));
                mnew[r] = fmaxf(mrow[r], vmx);
                const float alpha = __expf(mrow[r] - mnew[r]);
                mrow[r] = mnew[r];
                lrow[r] *= alpha;
#pragma unroll
                for (int c = 0; c < 4; ++c) acc[c][r] *= alpha;
            }
#pragma unroll
            for (int nt = 0; nt < 2; ++nt)
#pragma unroll
                for (int r = 0; r < 8; ++r) {
                    const float sv2 = sc[nt][r];
                    sc[nt][r] = (sv2 > 0.5f * NEGB) ? __expf(sv2 - mnew[r]) : 0.0f;
                }
#pragma unroll
            for (int r = 0; r < 8; ++r) {
                float rs = sc[0][r] + sc[1][r];
#pragma unroll
                for (int off = 1; off < 16; off <<= 1)
                    rs += __shfl_xor(rs, off, 32);
                lrow[r] += rs;
            }

            // ---- P: C/D layout -> A layout via per-wave LDS scratch ----
            _Float16* pw = &Ps[wid][0];
#pragma unroll
            for (int nt = 0; nt < 2; ++nt)
#pragma unroll
                for (int r = 0; r < 8; ++r)
                    pw[(base8 + r) * KT + nt * 16 + n16] = (_Float16)sc[nt][r];
            asm volatile("s_wait_dscnt 0x0" ::: "memory");  // same-wave RAW on LDS
            v16h pa;
            {
                const _Float16* pp = &pw[n16 * KT + base8];
                const v8h plo = *(const v8h*)pp;         // K = base8 + 0..7
                const v8h phi = *(const v8h*)(pp + 16);  // K = 16 + base8 + 0..7
                pa = cat8(plo, phi);
            }

            // ---- O += P V : 4 output d-chunks ----
#pragma unroll
            for (int c = 0; c < 4; ++c) {
                const _Float16* vp2 = &Vt[(c * 16 + n16) * KT + base8];
                const v8h blo = *(const v8h*)vp2;
                const v8h bhi = *(const v8h*)(vp2 + 16);
                acc[c] = __builtin_amdgcn_wmma_f32_16x16x32_f16(
                    false, pa, false, cat8(blo, bhi), (short)0, acc[c],
                    false, false);
            }
        }
        __syncthreads();  // protect Kl/Vt before next conversion
    }

    // ---------------- epilogue: O / l ----------------
    float* op = out + (((size_t)bb * S_ + qw + base8) * H_ + hh) * D_ + n16;
#pragma unroll
    for (int r = 0; r < 8; ++r) {
        const float invl = 1.0f / lrow[r];
#pragma unroll
        for (int c = 0; c < 4; ++c)
            op[(size_t)r * H_ * D_ + c * 16] = acc[c][r] * invl;
    }
}

extern "C" void kernel_launch(void* const* d_in, const int* in_sizes, int n_in,
                              void* d_out, int out_size, void* d_ws, size_t ws_size,
                              hipStream_t stream) {
    const float* q = (const float*)d_in[0];
    const float* k = (const float*)d_in[1];
    const float* v = (const float*)d_in[2];
    float* out = (float*)d_out;
    (void)in_sizes; (void)n_in; (void)out_size; (void)d_ws; (void)ws_size;
    dim3 grid(B_ * H_ * (S_ / QBLK));
    dim3 block(128);
    hipLaunchKernelGGL(fa_rope_swa_kernel, grid, block, 0, stream, q, k, v, out);
}
